// DEQPolicy_15496242004633
// MI455X (gfx1250) — compile-verified, matrix operand order checked by reading, not run on hardware
//
#include <hip/hip_runtime.h>

// ---------------------------------------------------------------------------
// DEQ policy network for MI455X (gfx1250, wave32, WMMA).
// Big GEMMs in bf16 via v_wmma_f32_16x16x32_bf16 (fp32 accum), operand tiles
// staged with CDNA5 async global->LDS DMA (ASYNCcnt). LayerNorm, Anderson
// acceleration and the tiny linear solves stay fp32.
// ---------------------------------------------------------------------------

typedef __bf16 bf16;
typedef __attribute__((ext_vector_type(8)))  __bf16 bf16x8;
typedef __attribute__((ext_vector_type(16))) __bf16 bf16x16;
typedef __attribute__((ext_vector_type(8)))  float  floatx8;

union BF16Frag { bf16x16 v; bf16x8 h[2]; };

#define BSZ   4096
#define NX    64
#define HDIM  1024
#define NOUT  2048      // NP*T
#define NP    32
#define MSLOT 5

// ---- CDNA5 async global->LDS copy (ASYNCcnt) ------------------------------
__device__ __forceinline__ void async_load_b128(unsigned lds_off, const void* gptr)
{
    asm volatile("global_load_async_to_lds_b128 %0, %1, off"
                 :: "v"(lds_off), "v"(gptr) : "memory");
}
__device__ __forceinline__ void wait_asynccnt0()
{
    asm volatile("s_wait_asynccnt 0" ::: "memory");
}
__device__ __forceinline__ unsigned lds_addr32(const void* p)
{
    return (unsigned)(unsigned long long)p;   // flat LDS addr low 32 bits = LDS offset
}

// ---------------------------------------------------------------------------
// GEMM: C[M,N] = A[M,K] * B[N,K]^T (+bias[n]) (opt relu) (+resid[m,n])
// A, B bf16 row-major with contiguous K. Block tile 128x128, BK=32, 8 waves;
// each wave owns a 32x64 patch = 2x4 WMMA 16x16 tiles (A-frag reused 4x).
// ---------------------------------------------------------------------------
__global__ __launch_bounds__(256) void gemm_bf16_wmma(
    const bf16* __restrict__ A, const bf16* __restrict__ B,
    const float* __restrict__ bias, const float* __restrict__ resid,
    float* __restrict__ outF, bf16* __restrict__ outBF,
    int M, int N, int K, int relu)
{
    __shared__ bf16 sA[128 * 32];
    __shared__ bf16 sB[128 * 32];

    const int tid  = threadIdx.x;
    const int lane = tid & 31;
    const int wave = tid >> 5;
    const int bm   = blockIdx.y * 128;
    const int bn   = blockIdx.x * 128;
    const int wm   = (wave >> 1) * 32;   // 0,32,64,96
    const int wn   = (wave & 1) * 64;    // 0,64

    const int lr = lane & 15;            // row/col within a 16-tile
    const int kh = (lane >> 4) * 8;      // lanes 16..31 take K+8 half

    const unsigned aBase = lds_addr32(&sA[0]);
    const unsigned bBase = lds_addr32(&sB[0]);

    floatx8 acc[2][4] = {};

    for (int k0 = 0; k0 < K; k0 += 32) {
        // ---- async DMA A tile (128x32) and B tile (128x32) into LDS ------
#pragma unroll
        for (int it = 0; it < 2; ++it) {
            int cc  = tid + it * 256;            // 512 chunks of 8 bf16 each
            int row = cc >> 2;
            int col = (cc & 3) * 8;
            unsigned soff = (unsigned)(row * 32 + col) * 2u;
            async_load_b128(aBase + soff, &A[(size_t)(bm + row) * K + k0 + col]);
            async_load_b128(bBase + soff, &B[(size_t)(bn + row) * K + k0 + col]);
        }
        wait_asynccnt0();
        __syncthreads();

        // ---- fragments per the CDNA5 16-bit A/B VGPR layout --------------
        BF16Frag af[2], bfr[4];
#pragma unroll
        for (int i = 0; i < 2; ++i) {
            int r = wm + 16 * i + lr;
            af[i].h[0] = *(const bf16x8*)&sA[r * 32 + kh];
            af[i].h[1] = *(const bf16x8*)&sA[r * 32 + 16 + kh];
        }
#pragma unroll
        for (int j = 0; j < 4; ++j) {
            int r = wn + 16 * j + lr;
            bfr[j].h[0] = *(const bf16x8*)&sB[r * 32 + kh];
            bfr[j].h[1] = *(const bf16x8*)&sB[r * 32 + 16 + kh];
        }
#pragma unroll
        for (int i = 0; i < 2; ++i)
#pragma unroll
            for (int j = 0; j < 4; ++j)
                acc[i][j] = __builtin_amdgcn_wmma_f32_16x16x32_bf16(
                    false, af[i].v, false, bfr[j].v,
                    (short)0, acc[i][j], false, false);
        __syncthreads();
    }

    // ---- epilogue: bias / relu / residual, fp32 and optional bf16 store --
    const int mofs = (lane >> 4) * 8;   // C/D layout: lanes 16..31 hold M+8
#pragma unroll
    for (int i = 0; i < 2; ++i) {
#pragma unroll
        for (int j = 0; j < 4; ++j) {
            int n = bn + wn + 16 * j + lr;
            float bv = bias ? bias[n] : 0.f;
#pragma unroll
            for (int r = 0; r < 8; ++r) {
                int m = bm + wm + 16 * i + mofs + r;
                float v = acc[i][j][r] + bv;
                if (relu) v = v > 0.f ? v : 0.f;
                size_t off = (size_t)m * N + n;
                if (resid) v += resid[off];
                if (outF)  outF[off] = v;
                if (outBF) outBF[off] = (bf16)v;
            }
        }
    }
}

// ---------------------------------------------------------------------------
// LayerNorm over rows of HDIM=1024. mode 0: y = LN(a); mode 1: y = LN(relu(a+b))
// outF written with row stride ostrF (lets us write straight into X/F slots).
// ---------------------------------------------------------------------------
__global__ __launch_bounds__(256) void ln_kernel(
    const float* __restrict__ a, const float* __restrict__ b,
    const float* __restrict__ g, const float* __restrict__ be,
    float* __restrict__ outF, int ostrF,
    bf16* __restrict__ outBF, int mode)
{
    const int row = blockIdx.x;
    const int tid = threadIdx.x;

    float v[4];
#pragma unroll
    for (int i = 0; i < 4; ++i) {
        int d = tid + i * 256;
        float x = a[(size_t)row * HDIM + d];
        if (mode == 1) {
            x += b[(size_t)row * HDIM + d];
            x = x > 0.f ? x : 0.f;
        }
        v[i] = x;
    }
    float s  = v[0] + v[1] + v[2] + v[3];
    float ss = v[0]*v[0] + v[1]*v[1] + v[2]*v[2] + v[3]*v[3];
    for (int o = 16; o; o >>= 1) { s += __shfl_down(s, o); ss += __shfl_down(ss, o); }

    __shared__ float rs[8], rss[8];
    int wave = tid >> 5, lane = tid & 31;
    if (lane == 0) { rs[wave] = s; rss[wave] = ss; }
    __syncthreads();
    if (tid == 0) {
        float ts = 0.f, tss = 0.f;
        for (int w = 0; w < 8; ++w) { ts += rs[w]; tss += rss[w]; }
        rs[0] = ts; rss[0] = tss;
    }
    __syncthreads();
    float mean = rs[0] * (1.f / HDIM);
    float var  = rss[0] * (1.f / HDIM) - mean * mean;
    float rstd = rsqrtf(var + 1e-5f);

#pragma unroll
    for (int i = 0; i < 4; ++i) {
        int d = tid + i * 256;
        float y = (v[i] - mean) * rstd * g[d] + be[d];
        if (outF)  outF[(size_t)row * ostrF + d] = y;
        if (outBF) outBF[(size_t)row * HDIM + d] = (bf16)y;
    }
}

// ---------------------------------------------------------------------------
// Anderson: per batch row compute GG = G G^T (G = F - X, n<=5 slots), build
// bordered system H (n+1)x(n+1), solve H a = e0, emit alpha = a[1:].
// One block of 256 threads per batch element.
// ---------------------------------------------------------------------------
__global__ __launch_bounds__(256) void anderson_alpha_kernel(
    const float* __restrict__ X, const float* __restrict__ F,
    float* __restrict__ alpha, int n)
{
    const int bidx = blockIdx.x;
    const int tid  = threadIdx.x;
    const float* Xb = X + (size_t)bidx * MSLOT * HDIM;
    const float* Fb = F + (size_t)bidx * MSLOT * HDIM;

    float p[15];
#pragma unroll
    for (int i = 0; i < 15; ++i) p[i] = 0.f;

    for (int d = tid; d < HDIM; d += 256) {
        float gv[5];
        for (int i = 0; i < n; ++i) gv[i] = Fb[i * HDIM + d] - Xb[i * HDIM + d];
        int c = 0;
        for (int i = 0; i < n; ++i)
            for (int j = i; j < n; ++j) p[c++] += gv[i] * gv[j];
    }

    __shared__ float red[15][8];
    int wave = tid >> 5, lane = tid & 31;
#pragma unroll
    for (int i = 0; i < 15; ++i) {
        float s = p[i];
        for (int o = 16; o; o >>= 1) s += __shfl_down(s, o);
        if (lane == 0) red[i][wave] = s;
    }
    __syncthreads();

    if (tid == 0) {
        float GG[5][5];
        int c = 0;
        for (int i = 0; i < n; ++i)
            for (int j = i; j < n; ++j) {
                float s = 0.f;
                for (int w = 0; w < 8; ++w) s += red[c][w];
                GG[i][j] = s; GG[j][i] = s; ++c;
            }
        // H a = e0, H = [[0,1...],[1,GG+lam*I]]
        float H[6][7];
        int nn = n + 1;
        for (int i = 0; i < nn; ++i)
            for (int j = 0; j <= nn; ++j) H[i][j] = 0.f;
        for (int j = 1; j < nn; ++j) { H[0][j] = 1.f; H[j][0] = 1.f; }
        for (int i = 1; i < nn; ++i)
            for (int j = 1; j < nn; ++j)
                H[i][j] = GG[i - 1][j - 1] + (i == j ? 1e-4f : 0.f);
        H[0][nn] = 1.f;  // rhs = e0
        // Gauss-Jordan with partial pivoting (H[0][0]==0 needs the pivot)
        for (int col = 0; col < nn; ++col) {
            int piv = col; float best = fabsf(H[col][col]);
            for (int r = col + 1; r < nn; ++r) {
                float t = fabsf(H[r][col]);
                if (t > best) { best = t; piv = r; }
            }
            if (piv != col)
                for (int j = col; j <= nn; ++j) {
                    float t = H[col][j]; H[col][j] = H[piv][j]; H[piv][j] = t;
                }
            float inv = 1.f / H[col][col];
            for (int j = col; j <= nn; ++j) H[col][j] *= inv;
            for (int r = 0; r < nn; ++r) {
                if (r == col) continue;
                float f = H[r][col];
                if (f != 0.f)
                    for (int j = col; j <= nn; ++j) H[r][j] -= f * H[col][j];
            }
        }
        for (int i = 0; i < n; ++i)
            alpha[(size_t)bidx * MSLOT + i] = H[i + 1][nn];
    }
}

// x_k = sum_i alpha_i * F_i  (beta = 1); store into X[:,slot] and bf16 z.
__global__ __launch_bounds__(256) void xk_kernel(
    const float* __restrict__ F, const float* __restrict__ alpha,
    float* __restrict__ X, bf16* __restrict__ zbf, int n, int slot)
{
    size_t idx = (size_t)blockIdx.x * 256 + threadIdx.x;
    int b = (int)(idx >> 10);
    int d = (int)(idx & 1023);
    const float* al = alpha + (size_t)b * MSLOT;
    const float* Fb = F + (size_t)b * MSLOT * HDIM;
    float s = 0.f;
    for (int i = 0; i < n; ++i) s += al[i] * Fb[i * HDIM + d];
    X[(size_t)b * MSLOT * HDIM + (size_t)slot * HDIM + d] = s;
    zbf[idx] = (bf16)s;
}

// ---------------------------------------------------------------------------
// small utility kernels
// ---------------------------------------------------------------------------
__global__ void cvt_f32_bf16(const float* __restrict__ in, bf16* __restrict__ o, int n)
{
    int i = blockIdx.x * 256 + threadIdx.x;
    if (i < n) o[i] = (bf16)in[i];
}
__global__ void zero_bf16_kernel(bf16* __restrict__ o, int n)
{
    int i = blockIdx.x * 256 + threadIdx.x;
    if (i < n) o[i] = (bf16)0.f;
}
__global__ void zero_xslot0_kernel(float* __restrict__ X)
{
    size_t idx = (size_t)blockIdx.x * 256 + threadIdx.x;   // BSZ*HDIM threads
    size_t b = idx >> 10, d = idx & 1023;
    X[b * MSLOT * HDIM + d] = 0.f;
}
__global__ void copy_x1_f0_kernel(float* __restrict__ X, const float* __restrict__ F)
{
    size_t idx = (size_t)blockIdx.x * 256 + threadIdx.x;
    size_t b = idx >> 10, d = idx & 1023;
    X[b * MSLOT * HDIM + HDIM + d] = F[b * MSLOT * HDIM + d];
}
__global__ void final_add_kernel(float* __restrict__ out, const float* __restrict__ x)
{
    size_t idx = (size_t)blockIdx.x * 256 + threadIdx.x;   // BSZ*NOUT threads
    int b  = (int)(idx >> 11);
    int np = (int)(idx & (NP - 1));
    out[idx] += x[(size_t)b * NX + np] * 10.f;
}

// ---------------------------------------------------------------------------
extern "C" void kernel_launch(void* const* d_in, const int* in_sizes, int n_in,
                              void* d_out, int out_size, void* d_ws, size_t ws_size,
                              hipStream_t stream)
{
    (void)in_sizes; (void)n_in; (void)out_size; (void)ws_size;

    const float* x      = (const float*)d_in[0];
    const float* W_inp  = (const float*)d_in[1];
    const float* b_inp  = (const float*)d_in[2];
    const float* g_inp  = (const float*)d_in[3];
    const float* be_inp = (const float*)d_in[4];
    const float* W1     = (const float*)d_in[5];
    const float* b1     = (const float*)d_in[6];
    const float* g1     = (const float*)d_in[7];
    const float* be1    = (const float*)d_in[8];
    const float* W2     = (const float*)d_in[9];
    const float* b2     = (const float*)d_in[10];
    const float* g2     = (const float*)d_in[11];
    const float* be2    = (const float*)d_in[12];
    const float* g3     = (const float*)d_in[13];
    const float* be3    = (const float*)d_in[14];
    const float* W_out  = (const float*)d_in[15];
    const float* b_out  = (const float*)d_in[16];
    float* out = (float*)d_out;

    // ---- carve workspace (~272 MB) --------------------------------------
    char* p = (char*)d_ws;
    auto carve = [&](size_t bytes) -> char* {
        char* r = p; p += (bytes + 255) & ~(size_t)255; return r;
    };
    bf16*  x_bf    = (bf16*) carve((size_t)BSZ * NX * 2);
    bf16*  Winp_bf = (bf16*) carve((size_t)HDIM * NX * 2);
    bf16*  W1_bf   = (bf16*) carve((size_t)HDIM * HDIM * 2);
    bf16*  W2_bf   = (bf16*) carve((size_t)HDIM * HDIM * 2);
    bf16*  Wout_bf = (bf16*) carve((size_t)NOUT * HDIM * 2);
    float* xinp    = (float*)carve((size_t)BSZ * HDIM * 4);
    bf16*  z_bf    = (bf16*) carve((size_t)BSZ * HDIM * 2);
    float* tmp     = (float*)carve((size_t)BSZ * HDIM * 4);
    float* z1      = (float*)carve((size_t)BSZ * HDIM * 4);
    bf16*  z1_bf   = (bf16*) carve((size_t)BSZ * HDIM * 2);
    float* inner   = (float*)carve((size_t)BSZ * HDIM * 4);
    float* zfin    = (float*)carve((size_t)BSZ * HDIM * 4);
    bf16*  zfin_bf = (bf16*) carve((size_t)BSZ * HDIM * 2);
    float* X       = (float*)carve((size_t)BSZ * MSLOT * HDIM * 4);
    float* F       = (float*)carve((size_t)BSZ * MSLOT * HDIM * 4);
    float* alpha   = (float*)carve((size_t)BSZ * MSLOT * 4);

    const int BH = BSZ * HDIM;

    auto gemm = [&](const bf16* A, const bf16* B, const float* bias,
                    const float* resid, float* oF, bf16* oBF,
                    int M, int N, int K, int relu) {
        dim3 g(N / 128, M / 128);
        gemm_bf16_wmma<<<g, 256, 0, stream>>>(A, B, bias, resid, oF, oBF, M, N, K, relu);
    };
    auto ln = [&](const float* a, const float* b, const float* g,
                  const float* be, float* oF, int ostr, bf16* oBF, int mode) {
        ln_kernel<<<BSZ, 256, 0, stream>>>(a, b, g, be, oF, ostr, oBF, mode);
    };
    // f(z): z1 = LN1(relu(z W1^T + b1)); inner = LN2(xinp + z1 W2^T + b2);
    //       out = LN3(relu(z1 + inner))
    auto frun = [&](float* outF, int ostr, bf16* outBF) {
        gemm(z_bf, W1_bf, b1, nullptr, tmp, nullptr, BSZ, HDIM, HDIM, 1);
        ln(tmp, nullptr, g1, be1, z1, HDIM, z1_bf, 0);
        gemm(z1_bf, W2_bf, b2, xinp, tmp, nullptr, BSZ, HDIM, HDIM, 0);
        ln(tmp, nullptr, g2, be2, inner, HDIM, nullptr, 0);
        ln(z1, inner, g3, be3, outF, ostr, outBF, 1);
    };

    // ---- one-time converts ----------------------------------------------
    cvt_f32_bf16<<<(BSZ * NX + 255) / 256, 256, 0, stream>>>(x, x_bf, BSZ * NX);
    cvt_f32_bf16<<<(HDIM * NX + 255) / 256, 256, 0, stream>>>(W_inp, Winp_bf, HDIM * NX);
    cvt_f32_bf16<<<(HDIM * HDIM + 255) / 256, 256, 0, stream>>>(W1, W1_bf, HDIM * HDIM);
    cvt_f32_bf16<<<(HDIM * HDIM + 255) / 256, 256, 0, stream>>>(W2, W2_bf, HDIM * HDIM);
    cvt_f32_bf16<<<(NOUT * HDIM + 255) / 256, 256, 0, stream>>>(W_out, Wout_bf, NOUT * HDIM);

    // ---- xinp = LN(x W_inp^T + b_inp) -----------------------------------
    gemm(x_bf, Winp_bf, b_inp, nullptr, tmp, nullptr, BSZ, HDIM, NX, 0);
    ln(tmp, nullptr, g_inp, be_inp, xinp, HDIM, nullptr, 0);

    // ---- Anderson init: X0 = 0, F0 = f(0), X1 = F0, F1 = f(F0) ----------
    zero_bf16_kernel<<<BH / 256, 256, 0, stream>>>(z_bf, BH);
    zero_xslot0_kernel<<<BH / 256, 256, 0, stream>>>(X);
    frun(F + 0 * HDIM, MSLOT * HDIM, z_bf);          // F0; z_bf <- bf16(F0)
    copy_x1_f0_kernel<<<BH / 256, 256, 0, stream>>>(X, F);
    frun(F + 1 * HDIM, MSLOT * HDIM, nullptr);       // F1 = f(F0)

    // ---- Anderson loop k = 2..14 ----------------------------------------
    for (int k = 2; k < 15; ++k) {
        int n    = k < MSLOT ? k : MSLOT;
        int slot = k % MSLOT;
        anderson_alpha_kernel<<<BSZ, 256, 0, stream>>>(X, F, alpha, n);
        xk_kernel<<<BH / 256, 256, 0, stream>>>(F, alpha, X, z_bf, n, slot);
        frun(F + slot * HDIM, MSLOT * HDIM, nullptr);
    }

    // z_star = X[:, 14%5] == last xk; z_bf already holds bf16(z_star).
    frun(zfin, HDIM, zfin_bf);                       // one extra f step

    // ---- output projection + reshape add --------------------------------
    gemm(zfin_bf, Wout_bf, b_out, nullptr, out, nullptr, BSZ, NOUT, HDIM, 0);
    final_add_kernel<<<(size_t)BSZ * NOUT / 256, 256, 0, stream>>>(out, x);
}
